// GCCN_2_63917703299194
// MI455X (gfx1250) — matrix operand order repaired and checked by reading, not compiled, not run on hardware
//
#include <hip/hip_runtime.h>

// Problem constants (match reference)
#define NV   20000
#define PB   5
#define TB   16
#define KN   3
#define INF  150
#define HF   64
#define OF   128
#define PT   (PB*TB)          // 80 bins
#define KTOT (PT*HF)          // 5120 contraction length
#define TILE_M 32             // vertices per block (2 x 16-row WMMA tiles)
#define NCHUNK 4
#define PTC  (PT/NCHUNK)      // 20 bins per LDS chunk
#define KC   (PTC*HF)         // 1280 halves per chunk
#define ROW_PAD (KC + 8)      // 1288 halves: stride = 644 dwords == 4 mod 64 -> conflict-free b128 reads

typedef __attribute__((ext_vector_type(16))) _Float16 v16h;
typedef __attribute__((ext_vector_type(8)))  _Float16 v8h;
typedef __attribute__((ext_vector_type(8)))  float    v8f;
typedef _Float16 half_t;

union FragU { int4 q[2]; v16h h; };   // WMMA operand (32 B)
union H8    { int4 q;    v8h  h; };   // 16 B of halves

// ---------------------------------------------------------------------------
// Kernel 1: Wg [O][H][P][T] f32  ->  Wg2 [O][(p*T+t)*H + h] f16  (B, K-major per column)
// ---------------------------------------------------------------------------
__global__ __launch_bounds__(256) void prep_wg(const float* __restrict__ wg,
                                               half_t* __restrict__ wg2) {
    int t = blockIdx.x * 256 + threadIdx.x;
    if (t >= OF * KTOT) return;
    int o  = t / KTOT;
    int k  = t - o * KTOT;
    int hi = k & (HF - 1);
    int pt = k >> 6;                   // p*T + t
    int p  = pt / TB;
    int tb = pt - p * TB;
    wg2[t] = (half_t)wg[((o * HF + hi) * PB + p) * TB + tb];
}

// ---------------------------------------------------------------------------
// Kernel 2: h = relu(x @ W1^T) stored as f16 [N,64]  (2.56 MB -> L2 resident)
// ---------------------------------------------------------------------------
__global__ __launch_bounds__(256) void linear_relu(const float* __restrict__ x,
                                                   const float* __restrict__ w1,
                                                   half_t* __restrict__ h16) {
    int t = blockIdx.x * 256 + threadIdx.x;
    if (t >= NV * HF) return;
    int c = t & (HF - 1);
    int n = t >> 6;
    const float* xr = x  + n * INF;
    const float* wr = w1 + c * INF;
    float s = 0.f;
#pragma unroll 5
    for (int i = 0; i < INF; ++i) s = fmaf(xr[i], wr[i], s);
    h16[t] = (half_t)(s > 0.f ? s : 0.f);
}

// ---------------------------------------------------------------------------
// Kernel 3: fused gather -> LDS patches (f16) -> WMMA GEMM -> bias -> L2 norm
// One block = 32 vertices, 256 threads = 8 waves; wave w owns output cols
// [16w,16w+16) for BOTH 16-row tiles (one B fragment feeds two WMMAs).
// ---------------------------------------------------------------------------
__global__ __launch_bounds__(256) void fused_patch_gemm(
        const half_t* __restrict__ h16,   // [N,64] f16 (L2 resident)
        const int*    __restrict__ cidx,  // [N,80,3]
        const float*  __restrict__ cw,    // [N,80,3]
        const half_t* __restrict__ wg2,   // [128,5120] f16
        const float*  __restrict__ bg,    // [128]
        float*        __restrict__ out)   // [N,128]
{
    extern __shared__ char smem[];
    half_t* patches = (half_t*)smem;                                      // 32 * ROW_PAD halves
    float*  outs    = (float*)(smem + TILE_M * ROW_PAD * sizeof(half_t)); // 32 x 128
    float*  rnorm   = outs + TILE_M * OF;                                 // 32

    const int nbase = blockIdx.x * TILE_M;
    const int tid   = threadIdx.x;
    const int lane  = tid & 31;
    const int wave  = tid >> 5;
    const int obase = wave * 16;

    v8f acc0 = {};   // rows 0..15
    v8f acc1 = {};   // rows 16..31

    for (int chunk = 0; chunk < NCHUNK; ++chunk) {
        __syncthreads();   // previous chunk's GEMM reads complete before overwrite
        // ---- Phase A: build 32 x (20*64) f16 patch chunk in LDS ----
        // 8 threads per patch vector, 8 columns each; packed-f16 weighted sum
        // (v_pk_mul/v_pk_fma), one ds_store_b128 per thread per slot.
        for (int s = tid; s < TILE_M * PTC * 8; s += 256) {
            int j = s & 7;                 // column group: cols 8j..8j+7
            int v = s >> 3;                // patch vector id
            int m = v & (TILE_M - 1);
            int b = v >> 5;                // local bin (v / TILE_M)
            int pt = chunk * PTC + b;
            int base = ((nbase + m) * PT + pt) * KN;
            int3   iv = *(const int3*)(cidx + base);
            float3 wv = *(const float3*)(cw + base);
            H8 g0, g1, g2, st;
            g0.q = *(const int4*)(h16 + (size_t)iv.x * HF + j * 8);
            g1.q = *(const int4*)(h16 + (size_t)iv.y * HF + j * 8);
            g2.q = *(const int4*)(h16 + (size_t)iv.z * HF + j * 8);
            half_t w0 = (half_t)wv.x, w1 = (half_t)wv.y, w2 = (half_t)wv.z;
            st.h = g0.h * w0 + g1.h * w1 + g2.h * w2;   // packed f16 FMAs
            *(int4*)(patches + m * ROW_PAD + b * HF + j * 8) = st.q;
        }
        __syncthreads();
        // ---- Phase B: 2x16-row GEMM over this chunk's K range ----
        const int mm     = lane & 15;      // A row / B column within tile
        const int hiHalf = lane >> 4;      // lane group 0 or 1
        const int c0     = hiHalf * 8;     // A: lanes<16 hold K 0..7 & 16..23
        const int kb     = hiHalf * 8;     // B: lanes<16 hold K 0..7 & 16..23
        const half_t* arow0 = patches + mm * ROW_PAD;
        const half_t* arow1 = patches + (mm + 16) * ROW_PAD;
        const half_t* brow  = wg2 + (size_t)(obase + mm) * KTOT + chunk * KC;
#pragma unroll 2
        for (int kk = 0; kk < KC; kk += 32) {
            FragU a0, a1, b;
            a0.q[0] = *(const int4*)(arow0 + kk + c0);       // ds_load_b128
            a0.q[1] = *(const int4*)(arow0 + kk + c0 + 16);
            a1.q[0] = *(const int4*)(arow1 + kk + c0);
            a1.q[1] = *(const int4*)(arow1 + kk + c0 + 16);
            b.q[0]  = *(const int4*)(brow + kk + kb);        // global_load_b128 (L2 hit)
            b.q[1]  = *(const int4*)(brow + kk + kb + 16);
            acc0 = __builtin_amdgcn_wmma_f32_16x16x32_f16(
                       false, a0.h, false, b.h, (short)0, acc0, false, false);
            acc1 = __builtin_amdgcn_wmma_f32_16x16x32_f16(
                       false, a1.h, false, b.h, (short)0, acc1, false, false);
        }
    }

    // ---- bias + stage C tiles to LDS (C layout: vgpr i -> M=i+8*(lane>=16), N=lane&15) ----
    {
        const int col  = obase + (lane & 15);
        const int mofs = (lane >> 4) * 8;
        float bias = bg[col];
#pragma unroll
        for (int i = 0; i < 8; ++i) {
            outs[(i + mofs) * OF + col]      = acc0[i] + bias;
            outs[(16 + i + mofs) * OF + col] = acc1[i] + bias;
        }
    }
    __syncthreads();
    // ---- per-row L2 norm ----
    if (tid < TILE_M) {
        float s = 0.f;
#pragma unroll 4
        for (int c = 0; c < OF; ++c) { float v = outs[tid * OF + c]; s += v * v; }
        rnorm[tid] = rsqrtf(s);
    }
    __syncthreads();
    // ---- normalized store ----
    for (int e = tid; e < TILE_M * OF; e += 256) {
        int r = e >> 7;
        int c = e & (OF - 1);
        out[(size_t)(nbase + r) * OF + c] = outs[r * OF + c] * rnorm[r];
    }
}

// ---------------------------------------------------------------------------
extern "C" void kernel_launch(void* const* d_in, const int* in_sizes, int n_in,
                              void* d_out, int out_size, void* d_ws, size_t ws_size,
                              hipStream_t stream) {
    const float* x    = (const float*)d_in[0];
    const int*   cidx = (const int*)  d_in[1];
    const float* cw   = (const float*)d_in[2];
    const float* w1   = (const float*)d_in[3];
    const float* wg   = (const float*)d_in[4];
    const float* bg   = (const float*)d_in[5];
    float* out = (float*)d_out;

    // workspace layout: h16 f16 [N,64] then Wg2 f16 [128,5120]
    half_t* h16 = (half_t*)d_ws;
    half_t* wg2 = (half_t*)((char*)d_ws + (size_t)NV * HF * sizeof(half_t)); // 2.56MB offset, 16B aligned

    prep_wg<<<(OF * KTOT + 255) / 256, 256, 0, stream>>>(wg, wg2);
    linear_relu<<<(NV * HF + 255) / 256, 256, 0, stream>>>(x, w1, h16);

    const size_t shmem = (size_t)TILE_M * ROW_PAD * sizeof(half_t)   // 82,432 B patches
                       + (size_t)TILE_M * OF * sizeof(float)         // 16,384 B C tiles
                       + (size_t)TILE_M * sizeof(float);             //    128 B norms
    fused_patch_gemm<<<NV / TILE_M, 256, shmem, stream>>>(h16, cidx, cw, wg2, bg, out);
}
